// GraphormerLayer_40492951667287
// MI455X (gfx1250) — matrix operand order repaired and checked by reading, not compiled, not run on hardware
//
#include <hip/hip_runtime.h>

// ---------------------------------------------------------------------------
// CDNA5 (gfx1250) Graphormer layer: WMMA f16 GEMMs + branchless flash attention.
// ---------------------------------------------------------------------------

typedef __attribute__((ext_vector_type(16))) _Float16 v16h;
typedef __attribute__((ext_vector_type(8)))  _Float16 v8h;
typedef __attribute__((ext_vector_type(8)))  float    v8f;

#define DEV __device__ __forceinline__

static constexpr int Nn    = 3072;
static constexpr int DIM   = 256;
static constexpr int FF    = 1024;
static constexpr int HEADS = 8;

DEV v8f wmma_f16(v16h a, v16h b, v8f c) {
  // D = A(16x32 f16) * B(32x16 f16) + C(16x16 f32)
  return __builtin_amdgcn_wmma_f32_16x16x32_f16(false, a, false, b, (short)0, c,
                                                false, false);
}

DEV v16h join16(v8h lo, v8h hi) {
  union { v16h v; v8h h[2]; } u;
  u.h[0] = lo; u.h[1] = hi;
  return u.v;
}

// A-fragment (16x32): lane l holds row (l&15), 16 consecutive K starting at
// kb + 16*(l>>4)  (contiguous order, matching the B-fragment).
DEV v16h load_a_frag(const _Float16* __restrict__ A, long lda, int m0, int kb, int lane) {
  const _Float16* p = A + (long)(m0 + (lane & 15)) * lda + kb + (lane >> 4) * 16;
  return join16(*(const v8h*)p, *(const v8h*)(p + 8));
}

// B-fragment (32x16) from Bt[n][k] (row n == column n of B): lane l holds
// column (l&15), 16 consecutive K starting at kb + 16*(l>>4).
DEV v16h load_b_frag(const _Float16* __restrict__ Bt, long ldb, int n0, int kb, int lane) {
  const _Float16* p = Bt + (long)(n0 + (lane & 15)) * ldb + kb + (lane >> 4) * 16;
  return join16(*(const v8h*)p, *(const v8h*)(p + 8));
}

DEV int clampi(int v, int lo, int hi) { return v < lo ? lo : (v > hi ? hi : v); }

// ---------------------------------------------------------------------------
// Prep kernels
// ---------------------------------------------------------------------------

template <typename T>
__global__ void transpose_to_f16_kernel(const T* __restrict__ in,
                                        _Float16* __restrict__ out,
                                        int R, int C) {
  long idx = (long)blockIdx.x * 256 + threadIdx.x;  // index into output [C][R]
  if (idx >= (long)R * C) return;
  int c = (int)(idx / R);
  int r = (int)(idx % R);
  out[idx] = (_Float16)(float)in[(long)r * C + c];
}

__global__ void add_cent_kernel(const float* __restrict__ x,
                                const int* __restrict__ cin,
                                const int* __restrict__ cout_,
                                const float* __restrict__ cinE,
                                const float* __restrict__ coutE,
                                _Float16* __restrict__ h16) {
  int i = blockIdx.x, d = threadIdx.x;
  int a = clampi(cin[i], 0, 50);
  int b = clampi(cout_[i], 0, 50);
  long o = (long)i * DIM + d;
  h16[o] = (_Float16)(x[o] + cinE[(long)a * DIM + d] + coutE[(long)b * DIM + d]);
}

// Fuse (spd, edge) -> 7-bit table index once, streaming: int4+int4 -> uchar4.
// Removes all clamp/index VALU and 7/8 of the graph-bias bytes from the
// attention hot loop.
__global__ void __launch_bounds__(256)
pack_idx_kernel(const int* __restrict__ spd, const int* __restrict__ edge,
                unsigned char* __restrict__ pk) {
  long t = ((long)blockIdx.x * 256 + threadIdx.x) * 4;
  int4 s = *(const int4*)(spd + t);
  int4 e = *(const int4*)(edge + t);
  uchar4 o;
  o.x = (unsigned char)(clampi(s.x, 0, 12) * 7 + clampi(e.x, 0, 6));
  o.y = (unsigned char)(clampi(s.y, 0, 12) * 7 + clampi(e.y, 0, 6));
  o.z = (unsigned char)(clampi(s.z, 0, 12) * 7 + clampi(e.z, 0, 6));
  o.w = (unsigned char)(clampi(s.w, 0, 12) * 7 + clampi(e.w, 0, 6));
  *(uchar4*)(pk + t) = o;
}

// ---------------------------------------------------------------------------
// WMMA GEMM: C[M,Nout] = A_f16[M,K] * Bt_f16[Nout,K]^T + bias  (+ epilogue)
// block = 256 threads (8 waves); wave tile 32x32; block tile 64(M) x 128(N)
// ---------------------------------------------------------------------------

enum { EPI_F16 = 0, EPI_GELU_F16 = 1, EPI_F32 = 2 };

template <int EPI>
DEV void store_tile(v8f acc, int mb, int nb, int hi, int c,
                    const float* __restrict__ bias, void* __restrict__ Cout, int Nout) {
  float bv = bias[nb + c];
#pragma unroll
  for (int r = 0; r < 8; ++r) {
    float v = acc[r] + bv;
    if (EPI == EPI_GELU_F16)
      v = 0.5f * v * (1.0f + erff(v * 0.70710678118654752f));  // exact gelu
    long off = (long)(mb + r + 8 * hi) * Nout + nb + c;
    if (EPI == EPI_F32) ((float*)Cout)[off] = v;
    else                ((_Float16*)Cout)[off] = (_Float16)v;
  }
}

template <int KDIM, int EPI>
__global__ void __launch_bounds__(256)
gemm_wmma_kernel(const _Float16* __restrict__ A, const _Float16* __restrict__ Bt,
                 const float* __restrict__ bias, void* __restrict__ Cout, int Nout) {
  const int lane = threadIdx.x & 31;
  const int wid  = threadIdx.x >> 5;
  const int m0 = blockIdx.x * 64  + (wid >> 2) * 32;
  const int n0 = blockIdx.y * 128 + (wid & 3) * 32;

  v8f acc00{}, acc01{}, acc10{}, acc11{};
#pragma unroll 2
  for (int kb = 0; kb < KDIM; kb += 32) {
    v16h a0 = load_a_frag(A, KDIM, m0,      kb, lane);
    v16h a1 = load_a_frag(A, KDIM, m0 + 16, kb, lane);
    v16h b0 = load_b_frag(Bt, KDIM, n0,      kb, lane);
    v16h b1 = load_b_frag(Bt, KDIM, n0 + 16, kb, lane);
    if (kb + 32 < KDIM) {
      __builtin_prefetch(A + (long)(m0 + (lane & 15)) * KDIM + kb + 32, 0, 0);
      __builtin_prefetch(Bt + (long)(n0 + (lane & 15)) * KDIM + kb + 32, 0, 0);
    }
    acc00 = wmma_f16(a0, b0, acc00);
    acc01 = wmma_f16(a0, b1, acc01);
    acc10 = wmma_f16(a1, b0, acc10);
    acc11 = wmma_f16(a1, b1, acc11);
  }

  const int hi = lane >> 4, c = lane & 15;
  store_tile<EPI>(acc00, m0,      n0,      hi, c, bias, Cout, Nout);
  store_tile<EPI>(acc01, m0,      n0 + 16, hi, c, bias, Cout, Nout);
  store_tile<EPI>(acc10, m0 + 16, n0,      hi, c, bias, Cout, Nout);
  store_tile<EPI>(acc11, m0 + 16, n0 + 16, hi, c, bias, Cout, Nout);
}

// ---------------------------------------------------------------------------
// Attention, one block = 16 query rows x all 8 heads (8 waves).  Wave w owns
// head w for rows i0..i0+15.  All waves touch the same pk rows at the same
// time -> WGP$/L2 hits; HBM side is ~9.4MB of packed indices.
//
// Branchless: per-head fused bias table ctab[91] in LDS carries
// spd_emb + edge_emb and bakes the spd>10 mask as -1e9 (exp -> 0).  Scores
// are O(1) for this model so exp() without running-max is exact in f32;
// denominator is a local accumulator reduced once at the end.
// ---------------------------------------------------------------------------

__global__ void __launch_bounds__(256)
attn_kernel(const _Float16* __restrict__ Q, const _Float16* __restrict__ Kf,
            const _Float16* __restrict__ Vt,          // Vt: [DIM][N]
            const unsigned char* __restrict__ pk,     // [N][N] packed spd*7+edge
            const float* __restrict__ spd_emb,        // [11][8]
            const float* __restrict__ edge_emb,       // [6][8]
            _Float16* __restrict__ Oout) {
  __shared__ _Float16 pbuf[8][16 * 32];
  __shared__ float    ctab[8][96];                    // [head][spd*7+edge], 91 used

  const int lane = threadIdx.x & 31, wid = threadIdx.x >> 5;
  const int i0 = blockIdx.x * 16;
  const int hi = lane >> 4, c = lane & 15;
  const int h  = wid;

  // Build fused (spd,edge) -> bias tables, one per head.  728 entries total.
  for (int t = threadIdx.x; t < 8 * 91; t += 256) {
    int hh = t / 91, se = t % 91;
    int s = se / 7, e = se % 7;
    int ec = e > 5 ? 5 : e;
    float v = (s <= 10) ? spd_emb[s * HEADS + hh] + edge_emb[ec * HEADS + hh]
                        : -1.0e9f;
    ctab[hh][se] = v;
  }
  __syncthreads();
  const float* ct = ctab[h];

  // Q A-fragment: 16 rows x 32 head dims, loaded once.
  const _Float16* qp = Q + (long)(i0 + c) * DIM + h * 32 + hi * 16;
  const v16h qf = join16(*(const v8h*)qp, *(const v8h*)(qp + 8));

  // Loop-invariant packed-index row pointers (one per r).
  const unsigned char* prow[8];
#pragma unroll
  for (int r = 0; r < 8; ++r) prow[r] = pk + (long)(i0 + r + 8 * hi) * Nn + c;

  v8f o0{}, o1{};
  float dacc[8];
#pragma unroll
  for (int r = 0; r < 8; ++r) dacc[r] = 0.0f;

  const float inv_s = 0.17677669529663687f;  // 1/sqrt(32)

  for (int j0 = 0; j0 < Nn; j0 += 32) {
    // K rows are already in B-fragment layout (row-major [N][DIM]).
    const _Float16* kp0 = Kf + (long)(j0 + c) * DIM + h * 32 + hi * 16;
    const _Float16* kp1 = Kf + (long)(j0 + 16 + c) * DIM + h * 32 + hi * 16;
    v16h k0 = join16(*(const v8h*)kp0, *(const v8h*)(kp0 + 8));
    v16h k1 = join16(*(const v8h*)kp1, *(const v8h*)(kp1 + 8));
    v8f s0{}, s1{};
    s0 = wmma_f16(qf, k0, s0);
    s1 = wmma_f16(qf, k1, s1);

    float p0[8], p1[8];
#pragma unroll
    for (int r = 0; r < 8; ++r) {
      int ia = prow[r][j0];                  // u8 -> fused bias index
      int ib = prow[r][j0 + 16];
      float pa = __expf(fmaf(s0[r], inv_s, ct[ia]));
      float pb = __expf(fmaf(s1[r], inv_s, ct[ib]));
      dacc[r] += pa + pb;
      p0[r] = pa;
      p1[r] = pb;
    }

    // Re-layout P: C-layout -> A-fragment layout through per-wave LDS tile.
    // DS ops from one wave execute in order, so no barrier is needed.
    _Float16* pb_ = &pbuf[wid][0];
#pragma unroll
    for (int r = 0; r < 8; ++r) {
      int row = r + 8 * hi;
      pb_[row * 32 + c]      = (_Float16)p0[r];
      pb_[row * 32 + 16 + c] = (_Float16)p1[r];
    }
    const _Float16* pl = pb_ + (lane & 15) * 32 + hi * 16;
    v16h pf = join16(*(const v8h*)pl, *(const v8h*)(pl + 8));

    // V B-fragments from Vt[DIM][N] (contiguous in j).
    const _Float16* vp0 = Vt + (long)(h * 32 + c) * Nn + j0 + hi * 16;
    const _Float16* vp1 = Vt + (long)(h * 32 + 16 + c) * Nn + j0 + hi * 16;
    v16h v0 = join16(*(const v8h*)vp0, *(const v8h*)(vp0 + 8));
    v16h v1 = join16(*(const v8h*)vp1, *(const v8h*)(vp1 + 8));

    o0 = wmma_f16(pf, v0, o0);
    o1 = wmma_f16(pf, v1, o1);
  }

  // One denominator reduction at the end (row lives in a 16-lane half).
#pragma unroll
  for (int r = 0; r < 8; ++r) {
    float d = dacc[r];
#pragma unroll
    for (int off = 1; off < 16; off <<= 1) d += __shfl_xor(d, off, 32);
    dacc[r] = d;
  }

#pragma unroll
  for (int r = 0; r < 8; ++r) {
    long i = i0 + r + 8 * hi;
    float il = 1.0f / dacc[r];
    Oout[i * DIM + h * 32 + c]      = (_Float16)(o0[r] * il);
    Oout[i * DIM + h * 32 + 16 + c] = (_Float16)(o1[r] * il);
  }
}

// ---------------------------------------------------------------------------
// y = LayerNorm(a + b) * g + be ; optional f16 copy.  One block per row.
// ---------------------------------------------------------------------------

__global__ void __launch_bounds__(256)
add_ln_kernel(const float* __restrict__ A, const float* __restrict__ B,
              const float* __restrict__ g, const float* __restrict__ be,
              float* __restrict__ outF, _Float16* __restrict__ outH) {
  __shared__ float sb[256];
  const int d = threadIdx.x;
  const long o = (long)blockIdx.x * DIM + d;
  float v = A[o] + B[o];
  sb[d] = v;
  __syncthreads();
#pragma unroll
  for (int s = 128; s > 0; s >>= 1) {
    if (d < s) sb[d] += sb[d + s];
    __syncthreads();
  }
  float mu = sb[0] * (1.0f / 256.0f);
  __syncthreads();
  float dv = v - mu;
  sb[d] = dv * dv;
  __syncthreads();
#pragma unroll
  for (int s = 128; s > 0; s >>= 1) {
    if (d < s) sb[d] += sb[d + s];
    __syncthreads();
  }
  float var = sb[0] * (1.0f / 256.0f);
  float y = dv * rsqrtf(var + 1e-5f) * g[d] + be[d];
  outF[o] = y;
  if (outH) outH[o] = (_Float16)y;
}

// ---------------------------------------------------------------------------
// Host orchestration
// ---------------------------------------------------------------------------

extern "C" void kernel_launch(void* const* d_in, const int* in_sizes, int n_in,
                              void* d_out, int out_size, void* d_ws, size_t ws_size,
                              hipStream_t stream) {
  const float* x        = (const float*)d_in[0];
  const int*   spd      = (const int*)d_in[1];
  const int*   cin      = (const int*)d_in[2];
  const int*   cout_    = (const int*)d_in[3];
  const int*   edge     = (const int*)d_in[4];
  const float* spd_emb  = (const float*)d_in[5];
  const float* cin_emb  = (const float*)d_in[6];
  const float* cout_emb = (const float*)d_in[7];
  const float* edge_emb = (const float*)d_in[8];
  const float* Wq = (const float*)d_in[9];  const float* bq = (const float*)d_in[10];
  const float* Wk = (const float*)d_in[11]; const float* bk = (const float*)d_in[12];
  const float* Wv = (const float*)d_in[13]; const float* bv = (const float*)d_in[14];
  const float* Wo = (const float*)d_in[15]; const float* bo = (const float*)d_in[16];
  const float* W1 = (const float*)d_in[17]; const float* b1 = (const float*)d_in[18];
  const float* W2 = (const float*)d_in[19]; const float* b2 = (const float*)d_in[20];
  const float* g1 = (const float*)d_in[21]; const float* be1 = (const float*)d_in[22];
  const float* g2 = (const float*)d_in[23]; const float* be2 = (const float*)d_in[24];

  char* w = (char*)d_ws;
  auto bump = [&](size_t bytes) -> char* {
    char* p = w;
    w += (bytes + 255) & ~(size_t)255;
    return p;
  };
  _Float16* h16   = (_Float16*)bump((size_t)Nn * DIM * 2);
  _Float16* WqT   = (_Float16*)bump((size_t)DIM * DIM * 2);
  _Float16* WkT   = (_Float16*)bump((size_t)DIM * DIM * 2);
  _Float16* WvT   = (_Float16*)bump((size_t)DIM * DIM * 2);
  _Float16* WoT   = (_Float16*)bump((size_t)DIM * DIM * 2);
  _Float16* W1T   = (_Float16*)bump((size_t)DIM * FF * 2);
  _Float16* W2T   = (_Float16*)bump((size_t)FF * DIM * 2);
  _Float16* Q16   = (_Float16*)bump((size_t)Nn * DIM * 2);
  _Float16* K16   = (_Float16*)bump((size_t)Nn * DIM * 2);
  _Float16* V16   = (_Float16*)bump((size_t)Nn * DIM * 2);
  _Float16* Vt16  = (_Float16*)bump((size_t)Nn * DIM * 2);
  _Float16* att16 = (_Float16*)bump((size_t)Nn * DIM * 2);
  float*    proj  = (float*)bump((size_t)Nn * DIM * 4);
  float*    x1f   = (float*)bump((size_t)Nn * DIM * 4);
  _Float16* x1h   = (_Float16*)bump((size_t)Nn * DIM * 2);
  _Float16* ff1   = (_Float16*)bump((size_t)Nn * FF * 2);
  float*    ff2   = (float*)bump((size_t)Nn * DIM * 4);
  unsigned char* pk = (unsigned char*)bump((size_t)Nn * Nn);

  // Packed graph-bias indices (streams the 75MB of spd/edge exactly once).
  pack_idx_kernel<<<(Nn * Nn / 4 + 255) / 256, 256, 0, stream>>>(spd, edge, pk);

  // Weight transposes (f32 -> f16, [K][N] -> [N][K])
  transpose_to_f16_kernel<float><<<(DIM * DIM + 255) / 256, 256, 0, stream>>>(Wq, WqT, DIM, DIM);
  transpose_to_f16_kernel<float><<<(DIM * DIM + 255) / 256, 256, 0, stream>>>(Wk, WkT, DIM, DIM);
  transpose_to_f16_kernel<float><<<(DIM * DIM + 255) / 256, 256, 0, stream>>>(Wv, WvT, DIM, DIM);
  transpose_to_f16_kernel<float><<<(DIM * DIM + 255) / 256, 256, 0, stream>>>(Wo, WoT, DIM, DIM);
  transpose_to_f16_kernel<float><<<(DIM * FF  + 255) / 256, 256, 0, stream>>>(W1, W1T, DIM, FF);
  transpose_to_f16_kernel<float><<<(FF  * DIM + 255) / 256, 256, 0, stream>>>(W2, W2T, FF, DIM);

  // h = x + cent_in_emb[cin] + cent_out_emb[cout]   (f16)
  add_cent_kernel<<<Nn, DIM, 0, stream>>>(x, cin, cout_, cin_emb, cout_emb, h16);

  // Q, K, V projections
  dim3 g256(Nn / 64, DIM / 128);
  gemm_wmma_kernel<DIM, EPI_F16><<<g256, 256, 0, stream>>>(h16, WqT, bq, Q16, DIM);
  gemm_wmma_kernel<DIM, EPI_F16><<<g256, 256, 0, stream>>>(h16, WkT, bk, K16, DIM);
  gemm_wmma_kernel<DIM, EPI_F16><<<g256, 256, 0, stream>>>(h16, WvT, bv, V16, DIM);

  // Vt[DIM][N] for contiguous P.V B-fragments
  transpose_to_f16_kernel<_Float16><<<(Nn * DIM + 255) / 256, 256, 0, stream>>>(V16, Vt16, Nn, DIM);

  // Attention: one block per 16-row i-tile, 8 waves = 8 heads.
  attn_kernel<<<Nn / 16, 256, 0, stream>>>(Q16, K16, Vt16, pk,
                                           spd_emb, edge_emb, att16);

  // Output projection, residual + LN1
  gemm_wmma_kernel<DIM, EPI_F32><<<g256, 256, 0, stream>>>(att16, WoT, bo, proj, DIM);
  add_ln_kernel<<<Nn, DIM, 0, stream>>>(x, proj, g1, be1, x1f, x1h);

  // FFN: gelu(x1 @ W1 + b1) @ W2 + b2, residual + LN2 -> d_out
  gemm_wmma_kernel<DIM, EPI_GELU_F16><<<dim3(Nn / 64, FF / 128), 256, 0, stream>>>(x1h, W1T, b1, ff1, FF);
  gemm_wmma_kernel<FF, EPI_F32><<<g256, 256, 0, stream>>>(ff1, W2T, b2, ff2, DIM);
  add_ln_kernel<<<Nn, DIM, 0, stream>>>(x1f, ff2, g2, be2, (float*)d_out, nullptr);

  (void)in_sizes; (void)n_in; (void)out_size; (void)ws_size;
}
